// GRUDModel_6107443495375
// MI455X (gfx1250) — compile-verified, matrix operand order checked by reading, not run on hardware
//
#include <hip/hip_runtime.h>
#include <hip/hip_bf16.h>
#include <math.h>

#define B_  128
#define L_  512
#define DI_ 128
#define DM_ 512
#define G3_ (3*DM_)   // 1536
#define BT_ 16        // batch rows per batch-group (WMMA M)
#define NCG_ 8        // col-groups (workgroups per cluster), 1536/8 = 192 cols each
#define NTHR_ 128     // 4 waves: one 16-col d-subtile per wave

typedef __bf16 bf16;
typedef __attribute__((ext_vector_type(8)))  bf16  v8bf;
typedef __attribute__((ext_vector_type(16))) bf16  v16bf;
typedef __attribute__((ext_vector_type(8)))  float v8f;

// ---------------------------------------------------------------------------
// Fragment loader matching the CDNA5 16-bit A/B WMMA layout:
// lanes 0-15 hold rows 0-15 with K = {k..k+7, k+16..k+23};
// lanes 16-31 hold rows 0-15 with K = {k+8..k+15, k+24..k+31}.
// Two 16-byte loads per lane per K=32 slab (row-major source, stride ld).
// Works for both LDS and global (L2-resident) sources.
// ---------------------------------------------------------------------------
__device__ __forceinline__ v16bf load_frag(const bf16* __restrict__ base, int ld,
                                           int k, int lane) {
  int r  = lane & 15;
  int k0 = k + ((lane & 16) ? 8 : 0);
  v8bf lo = *(const v8bf*)(base + (size_t)r * ld + k0);
  v8bf hi = *(const v8bf*)(base + (size_t)r * ld + k0 + 16);
  return __builtin_shufflevector(lo, hi, 0,1,2,3,4,5,6,7,8,9,10,11,12,13,14,15);
}

__device__ __forceinline__ v8f wmma_bf16(v16bf a, v16bf b, v8f c) {
  return __builtin_amdgcn_wmma_f32_16x16x32_bf16(false, a, false, b,
                                                 (short)0, c, false, false);
}

__device__ __forceinline__ float sigm_(float x) { return 1.0f / (1.0f + __expf(-x)); }

// ---------------------------------------------------------------------------
// One GRU cell, one wave, one 16-col d-subtile (tiles dt, dt+32, dt+64).
// gi = actI @ Wi^T (K=KA), gh = actH @ Wh^T (K=DM).  r/z accumulate both
// phases into one chain; n keeps gi/gh separate (n = tanh(inn + r*hn)).
// Gates fuse in registers; hreg (WMMA C layout) is the recurrent state.
// Per K-slab: one A fragment feeds three independent WMMAs (r/z/n).
// ---------------------------------------------------------------------------
__device__ __forceinline__ void cell_tile(const bf16* __restrict__ actI, int KA,
                                          const bf16* __restrict__ actH,
                                          const bf16* __restrict__ Wi,
                                          const bf16* __restrict__ Wh,
                                          float brz, float bz, float bin, float bhn,
                                          v8f& hreg, int dt, int lane) {
  const bf16* Wir = Wi + (size_t)(dt     ) * 16 * KA;
  const bf16* Wiz = Wi + (size_t)(dt + 32) * 16 * KA;
  const bf16* Win = Wi + (size_t)(dt + 64) * 16 * KA;
  const bf16* Whr = Wh + (size_t)(dt     ) * 16 * DM_;
  const bf16* Whz = Wh + (size_t)(dt + 32) * 16 * DM_;
  const bf16* Whn = Wh + (size_t)(dt + 64) * 16 * DM_;
  v8f ar = {}, az = {}, ani = {}, anh = {};
  for (int k = 0; k < KA; k += 32) {
    v16bf a = load_frag(actI, KA, k, lane);
    ar  = wmma_bf16(a, load_frag(Wir, KA, k, lane), ar);
    az  = wmma_bf16(a, load_frag(Wiz, KA, k, lane), az);
    ani = wmma_bf16(a, load_frag(Win, KA, k, lane), ani);
  }
  for (int k = 0; k < DM_; k += 32) {
    v16bf a = load_frag(actH, DM_, k, lane);
    ar  = wmma_bf16(a, load_frag(Whr, DM_, k, lane), ar);
    az  = wmma_bf16(a, load_frag(Whz, DM_, k, lane), az);
    anh = wmma_bf16(a, load_frag(Whn, DM_, k, lane), anh);
  }
  for (int v = 0; v < 8; ++v) {
    float r = sigm_(ar[v] + brz);
    float z = sigm_(az[v] + bz);
    float n = tanhf(ani[v] + bin + r * (anh[v] + bhn));
    hreg[v] = (1.f - z) * n + z * hreg[v];
  }
}

// ---------------------------------------------------------------------------
// Cluster-cooperative GRU-D scan.
// Grid = 8 batch-groups x 8 col-groups.  The 8 col-groups of a batch-group
// form one (conceptual) workgroup cluster: each streams only 1/8 of the
// weights per step (the L2-BW bottleneck) and they exchange the 16KB hidden
// state through L2 ping-pong buffers, synchronized by cluster barriers.
// ---------------------------------------------------------------------------
__global__ __launch_bounds__(NTHR_)
void grud_scan(const float* __restrict__ x, const float* __restrict__ tau,
               const float* __restrict__ mask, const float* __restrict__ Wgx,
               const float* __restrict__ Wghw,
               const bf16* __restrict__ Wih0, const bf16* __restrict__ Whh0,
               const float* __restrict__ bih0, const float* __restrict__ bhh0,
               const bf16* __restrict__ Wih1, const bf16* __restrict__ Whh1,
               const float* __restrict__ bih1, const float* __restrict__ bhh1,
               bf16* __restrict__ h0x,   // [8 bg][2 parity][16*512]
               bf16* __restrict__ h1x,   // [8 bg][2 parity][16*512]
               bf16* __restrict__ seqbf) {
  __shared__ bf16  inpbf[BT_ * 2 * DI_];   // [x_imp | mask] 16x256
  __shared__ float xprev[BT_ * DI_];
  __shared__ float xm  [BT_ * DI_];

  const int tid  = threadIdx.x;
  const int lane = tid & 31;
  const int wave = tid >> 5;
  const int bg   = blockIdx.x >> 3;        // batch group
  const int cg   = blockIdx.x & 7;         // col group (cluster member)
  const int b0   = bg * BT_;
  const int dt   = cg * 4 + wave;          // d-subtile 0..31 (16 cols each)
  const int dcol = dt * 16 + (lane & 15);
  const int mrow = (lane >> 4) * 8;        // C layout: VGPR v -> M = mrow+v
  const int EX   = BT_ * DM_;

  bf16* h0buf = h0x + (size_t)bg * 2 * EX;
  bf16* h1buf = h1x + (size_t)bg * 2 * EX;

  // t-invariant per-lane biases (r/z fold gi+gh biases; n keeps them split)
  const float brz0 = bih0[dcol] + bhh0[dcol];
  const float bz0  = bih0[DM_ + dcol] + bhh0[DM_ + dcol];
  const float bin0 = bih0[2 * DM_ + dcol], bhn0 = bhh0[2 * DM_ + dcol];
  const float brz1 = bih1[dcol] + bhh1[dcol];
  const float bz1  = bih1[DM_ + dcol] + bhh1[DM_ + dcol];
  const float bin1 = bih1[2 * DM_ + dcol], bhn1 = bhh1[2 * DM_ + dcol];

  v8f h0reg = {}, h1reg = {};              // recurrent state in WMMA C layout

  // masked mean xm (redundant per col-group; reads shared via L2)
  for (int i = tid; i < BT_ * DI_; i += NTHR_) {
    int m = i / DI_, f = i % DI_;
    xprev[i] = 0.f;
    const float* xp = x    + ((size_t)(b0 + m) * L_) * DI_ + f;
    const float* mp = mask + ((size_t)(b0 + m) * L_) * DI_ + f;
    float sx = 0.f, sm = 0.f;
    for (int t = 0; t < L_; ++t) {
      float xv = xp[(size_t)t * DI_];
      if (xv != xv) xv = 0.f;              // nan_to_num
      float mv = mp[(size_t)t * DI_];
      sx += xv * mv; sm += mv;
    }
    xm[i] = sx / fmaxf(sm, 1.0f);
  }
  __syncthreads();

  for (int t = 0; t < L_; ++t) {
    // Opaque memory barrier: forbid LICM hoisting of t-invariant weight
    // fragment loads (would otherwise spill hundreds of VGPRs to scratch).
    asm volatile("" ::: "memory");
    const int p = t & 1;

    // (A) decayed imputation -> inpbf (LDS); decay h1 in registers and
    //     publish the decayed slice to h1buf[p].
    for (int i = tid; i < BT_ * DI_; i += NTHR_) {
      int m = i / DI_, f = i % DI_;
      float dtm = tau[(size_t)(b0 + m) * L_ + t];
      float gx  = __expf(-fmaxf(Wgx[f], 0.f) * dtm);
      float xv  = x[((size_t)(b0 + m) * L_ + t) * DI_ + f];
      if (xv != xv) xv = 0.f;
      float mt  = mask[((size_t)(b0 + m) * L_ + t) * DI_ + f];
      float xi  = mt * xv + (1.f - mt) * (gx * xprev[i] + (1.f - gx) * xm[i]);
      xprev[i] = xi;
      inpbf[m * (2 * DI_) + f]       = (bf16)xi;
      inpbf[m * (2 * DI_) + DI_ + f] = (bf16)mt;
    }
    bf16* h1wr = h1buf + p * EX;
    for (int v = 0; v < 8; ++v) {
      int m = mrow + v;
      float dtm = tau[(size_t)(b0 + m) * L_ + t];
      float g   = __expf(-fmaxf(dtm * Wghw[dcol], 0.f));
      h1reg[v] *= g;                       // h1 <- decayed h1
      h1wr[(size_t)m * DM_ + dcol] = (bf16)h1reg[v];
    }
    __syncthreads();                       // inpbf ready (workgroup-local)
    __threadfence();                       // release h1d slice to L2
    __builtin_amdgcn_s_cluster_barrier();  // B1: h1d ready cluster-wide
    __threadfence();                       // acquire

    // (C) cell 0: gi = inp@Wih0^T (K=256, LDS A), gh = h0@Whh0^T (K=512,
    //     A streamed from cluster-shared h0buf[p]); gates in registers.
    const bf16* h0rd = h0buf + p * EX;
    cell_tile(inpbf, 2 * DI_, h0rd, Wih0, Whh0,
              brz0, bz0, bin0, bhn0, h0reg, dt, lane);
    bf16* h0wr = h0buf + (1 - p) * EX;     // ping-pong: no read/write overlap
    for (int v = 0; v < 8; ++v)
      h0wr[(size_t)(mrow + v) * DM_ + dcol] = (bf16)h0reg[v];
    __threadfence();
    __builtin_amdgcn_s_cluster_barrier();  // B2: h0n ready cluster-wide
    __threadfence();

    // (D) cell 1: gi = h0n@Wih1^T, gh = h1d@Whh1^T (A from L2 exchange).
    cell_tile(h0wr, DM_, h1wr, Wih1, Whh1,
              brz1, bz1, bin1, bhn1, h1reg, dt, lane);
    for (int v = 0; v < 8; ++v)
      seqbf[(((size_t)(b0 + mrow + v)) * L_ + t) * DM_ + dcol] = (bf16)h1reg[v];
  }
}

// ---------------------------------------------------------------------------
// Attention scores: s[b,t] = valid ? tanh(seq@pW^T+pb)@pu^T : -inf
// ---------------------------------------------------------------------------
__global__ void attn_score(const bf16* __restrict__ seqbf, const float* __restrict__ mask,
                           const float* __restrict__ pW, const float* __restrict__ pb,
                           const float* __restrict__ pu, float* __restrict__ s) {
  int b = blockIdx.x, j = threadIdx.x;         // 256 threads = DM/2 proj units
  __shared__ float rsum[256];
  __shared__ float rmax[256];
  for (int t = 0; t < L_; ++t) {
    const bf16* sp = seqbf + ((size_t)b * L_ + t) * DM_;
    const float* wr = pW + (size_t)j * DM_;
    float acc = 0.f;
    for (int k = 0; k < DM_; ++k) acc += (float)sp[k] * wr[k];
    float pj = tanhf(acc + pb[j]) * pu[j];
    float vm = (j < DI_) ? mask[((size_t)b * L_ + t) * DI_ + j] : 0.f;
    rsum[j] = pj; rmax[j] = vm;
    __syncthreads();
    for (int off = 128; off > 0; off >>= 1) {
      if (j < off) { rsum[j] += rsum[j + off]; rmax[j] = fmaxf(rmax[j], rmax[j + off]); }
      __syncthreads();
    }
    if (j == 0) s[(size_t)b * L_ + t] = (rmax[0] > 0.f) ? rsum[0] : -INFINITY;
    __syncthreads();
  }
}

// ---------------------------------------------------------------------------
// Softmax over L, weighted pool, final linears.  out = [mort(128x2) | los(128)]
// ---------------------------------------------------------------------------
__global__ void attn_pool(const bf16* __restrict__ seqbf, const float* __restrict__ s,
                          const float* __restrict__ mw, const float* __restrict__ mb,
                          const float* __restrict__ lw, const float* __restrict__ lb,
                          float* __restrict__ out) {
  int b = blockIdx.x, tid = threadIdx.x;       // 512 threads (L_ == DM_)
  __shared__ float alpha[L_];
  __shared__ float red[DM_];
  float sv = s[(size_t)b * L_ + tid];
  red[tid] = sv;
  __syncthreads();
  for (int off = DM_ / 2; off > 0; off >>= 1) {
    if (tid < off) red[tid] = fmaxf(red[tid], red[tid + off]);
    __syncthreads();
  }
  float smax = red[0];
  __syncthreads();
  bool allinv = !(smax > -3.0e38f);            // all timesteps invalid -> uniform
  float e = allinv ? 1.f : __expf(sv - smax);
  red[tid] = e;
  __syncthreads();
  for (int off = DM_ / 2; off > 0; off >>= 1) {
    if (tid < off) red[tid] += red[tid + off];
    __syncthreads();
  }
  float denom = red[0];
  __syncthreads();
  alpha[tid] = e / denom;
  __syncthreads();

  float zd = 0.f;
  const bf16* sp = seqbf + ((size_t)b * L_) * DM_ + tid;
  for (int t = 0; t < L_; ++t) zd += alpha[t] * (float)sp[(size_t)t * DM_];

  float res[3];
  const float* wv[3] = { mw, mw + DM_, lw };
  for (int r = 0; r < 3; ++r) {
    red[tid] = zd * wv[r][tid];
    __syncthreads();
    for (int off = DM_ / 2; off > 0; off >>= 1) {
      if (tid < off) red[tid] += red[tid + off];
      __syncthreads();
    }
    res[r] = red[0];
    __syncthreads();
  }
  if (tid == 0) {
    out[b * 2 + 0]  = res[0] + mb[0];
    out[b * 2 + 1]  = res[1] + mb[1];
    out[2 * B_ + b] = res[2] + lb[0];
  }
}

__global__ void cvt_bf16(const float* __restrict__ src, bf16* __restrict__ dst, int n) {
  int i = blockIdx.x * blockDim.x + threadIdx.x;
  if (i < n) dst[i] = (bf16)src[i];
}

__global__ void zero_bf16(bf16* __restrict__ dst, int n) {
  int i = blockIdx.x * blockDim.x + threadIdx.x;
  if (i < n) dst[i] = (bf16)0.f;
}

// ---------------------------------------------------------------------------
extern "C" void kernel_launch(void* const* d_in, const int* in_sizes, int n_in,
                              void* d_out, int out_size, void* d_ws, size_t ws_size,
                              hipStream_t stream) {
  const float* x    = (const float*)d_in[0];
  const float* tau  = (const float*)d_in[1];
  const float* mask = (const float*)d_in[2];
  const float* Wgx  = (const float*)d_in[3];
  const float* Wghw = (const float*)d_in[4];
  const float* Wih0 = (const float*)d_in[5];
  const float* Whh0 = (const float*)d_in[6];
  const float* bih0 = (const float*)d_in[7];
  const float* bhh0 = (const float*)d_in[8];
  const float* Wih1 = (const float*)d_in[9];
  const float* Whh1 = (const float*)d_in[10];
  const float* bih1 = (const float*)d_in[11];
  const float* bhh1 = (const float*)d_in[12];
  const float* pW   = (const float*)d_in[13];
  const float* pb   = (const float*)d_in[14];
  const float* pu   = (const float*)d_in[15];
  const float* mw   = (const float*)d_in[16];
  const float* mb   = (const float*)d_in[17];
  const float* lw   = (const float*)d_in[18];
  const float* lb   = (const float*)d_in[19];
  float* out = (float*)d_out;

  char* ws = (char*)d_ws;
  bf16* Wih0b = (bf16*)ws; ws += (size_t)G3_ * (2 * DI_) * sizeof(bf16);
  bf16* Whh0b = (bf16*)ws; ws += (size_t)G3_ * DM_ * sizeof(bf16);
  bf16* Wih1b = (bf16*)ws; ws += (size_t)G3_ * DM_ * sizeof(bf16);
  bf16* Whh1b = (bf16*)ws; ws += (size_t)G3_ * DM_ * sizeof(bf16);
  bf16* h0x   = (bf16*)ws; ws += (size_t)8 * 2 * BT_ * DM_ * sizeof(bf16); // 256 KB
  bf16* h1x   = (bf16*)ws; ws += (size_t)8 * 2 * BT_ * DM_ * sizeof(bf16); // 256 KB
  bf16* seqb  = (bf16*)ws; ws += (size_t)B_ * L_ * DM_ * sizeof(bf16);
  float* scores = (float*)ws; ws += (size_t)B_ * L_ * sizeof(float);

  int n0 = G3_ * 2 * DI_, n1 = G3_ * DM_;
  cvt_bf16<<<(n0 + 255) / 256, 256, 0, stream>>>(Wih0, Wih0b, n0);
  cvt_bf16<<<(n1 + 255) / 256, 256, 0, stream>>>(Whh0, Whh0b, n1);
  cvt_bf16<<<(n1 + 255) / 256, 256, 0, stream>>>(Wih1, Wih1b, n1);
  cvt_bf16<<<(n1 + 255) / 256, 256, 0, stream>>>(Whh1, Whh1b, n1);
  int nz = 2 * 8 * 2 * BT_ * DM_;              // both exchange buffers
  zero_bf16<<<(nz + 255) / 256, 256, 0, stream>>>(h0x, nz);

  // 8 batch-groups x 8 col-groups; the 8 col-groups of a batch-group form a
  // workgroup cluster (dispatch-configured on hardware; barrier degrades to
  // NOP outside clusters per ISA 3.2).
  grud_scan<<<8 * NCG_, NTHR_, 0, stream>>>(x, tau, mask, Wgx, Wghw,
                                            Wih0b, Whh0b, bih0, bhh0,
                                            Wih1b, Whh1b, bih1, bhh1,
                                            h0x, h1x, seqb);

  attn_score<<<B_, 256, 0, stream>>>(seqb, mask, pW, pb, pu, scores);
  attn_pool<<<B_, DM_, 0, stream>>>(seqb, scores, mw, mb, lw, lb, out);
}
